// DimeNetPlusPlus_7275674599724
// MI455X (gfx1250) — compile-verified, matrix operand order checked by reading, not run on hardware
//
#include <hip/hip_runtime.h>
#include <hip/hip_bf16.h>
#include <cstddef>

// ---------------------------------------------------------------------------
// DimeNet++ forward for MI455X (gfx1250, wave32, WMMA).
// All 128x128 GEMMs run on v_wmma_f32_16x16x32_f16 (f16 in, f32 acc).
// Weights staged into LDS fragment-major (one aligned 32B read per B frag).
// Uniform full-tile fast paths avoid per-element EXEC-mask store guards.
// ---------------------------------------------------------------------------

typedef __attribute__((ext_vector_type(16))) _Float16 v16h;
typedef __attribute__((ext_vector_type(8)))  _Float16 v8h;
typedef __attribute__((ext_vector_type(8)))  float    v8f;

#define HID 128
#define NR  16
#define PI_F 3.14159265358979323846f

__device__ __forceinline__ float silu_f(float x) { return x / (1.f + __expf(-x)); }
__device__ __forceinline__ float sigm_f(float x) { return 1.f / (1.f + __expf(-x)); }

// K offset inside a 16-element f16 fragment (ISA 7.12.2, 16-bit A 16x32 layout).
__device__ __forceinline__ int frag_k(int e, int half) {
    int p = e >> 1, slot = e & 1;
    return ((p >> 2) << 4) + ((p & 3) << 1) + (half << 3) + slot;
}

// A fragment from a row-major LDS tile: two contiguous 8-half (16 B) runs.
__device__ __forceinline__ v16h load_a(const _Float16* As, int lda, int mBase, int kBase,
                                       int l16, int half) {
    const _Float16* p = As + (mBase + l16) * lda + kBase + 8 * half;
    union { v16h v; struct { v8h lo; v8h hi; } s; } u;
    u.s.lo = *(const v8h*)(p);
    u.s.hi = *(const v8h*)(p + 16);
    return u.v;
}

// Stage a 128x128 row-major fp32 weight matrix into LDS, swizzled so that the
// fragment for (k-tile kt, n-tile nt, lane) is 16 contiguous f16 (32 B).
__device__ __forceinline__ void stage_w_sw(const float* __restrict__ w, _Float16* s_w, int tid) {
    for (int ent = tid; ent < 4 * 8 * 32; ent += 256) {   // kt*256 + nt*32 + lane
        int lane = ent & 31;
        int nt   = (ent >> 5) & 7;
        int kt   = ent >> 8;
        int half = lane >> 4, no = lane & 15;
        _Float16* dstp = s_w + ent * 16;
#pragma unroll
        for (int e = 0; e < 16; ++e) {
            int k = kt * 32 + frag_k(e, half);
            dstp[e] = (_Float16)w[k * HID + nt * 16 + no];
        }
    }
}

// B fragment: one aligned 32-byte LDS vector read.
__device__ __forceinline__ v16h load_b_sw(const _Float16* Bs, int kt, int nt, int lane) {
    return *(const v16h*)(Bs + ((kt * 8 + nt) * 32 + lane) * 16);
}

__device__ __forceinline__ v8f wmma_f16(v16h a, v16h b, v8f c) {
    return __builtin_amdgcn_wmma_f32_16x16x32_f16(false, a, false, b, (short)0, c, false, false);
}

// ---------------------------------------------------------------------------
// Setup kernels
// ---------------------------------------------------------------------------

__global__ void k_rbf(const float* __restrict__ d, float* __restrict__ rbf, int E) {
    int e = blockIdx.x * blockDim.x + threadIdx.x;
    if (e >= E) return;
    float dd = d[e];
    float x  = dd * (1.f / 5.0f);
    float x2 = x * x;
    float x6 = x2 * x2 * x2;
    float env = (1.f - 28.f * x6 + 48.f * x6 * x - 21.f * x6 * x2) * (dd < 5.0f ? 1.f : 0.f);
    float s = sqrtf(2.f / 5.0f) / dd * env;
#pragma unroll
    for (int n = 0; n < NR; ++n)
        rbf[(size_t)e * NR + n] = s * __sinf((float)(n + 1) * (PI_F / 5.0f) * dd);
}

// column-means of ib_sph_w[blk] (7x8) and mean of ib_sph_b[blk] (8)
__global__ void k_sphmean(const float* __restrict__ sw, const float* __restrict__ sb,
                          float* __restrict__ sphm) {
    int tid = threadIdx.x;
    if (tid >= 32) return;
    int blk = tid >> 3, q = tid & 7;
    float acc = 0.f;
    if (q < 7) {
#pragma unroll
        for (int j = 0; j < 8; ++j) acc += sw[blk * 56 + q * 8 + j];
    } else {
#pragma unroll
        for (int j = 0; j < 8; ++j) acc += sb[blk * 8 + j];
    }
    sphm[blk * 8 + q] = acc * 0.125f;
}

__global__ void k_angle(const float* __restrict__ dirs, const float* __restrict__ sphm,
                        float* __restrict__ angle, int E) {
    int e = blockIdx.x * blockDim.x + threadIdx.x;
    if (e >= E) return;
    float x = dirs[e * 3 + 0];
    float y = dirs[e * 3 + 1];
    float z = dirs[e * 3 + 2];
    float f[7] = {1.f, y, z, x, x * y, y * z, 3.f * z * z - 1.f};
#pragma unroll
    for (int blk = 0; blk < 4; ++blk) {
        float acc = sphm[blk * 8 + 7];
#pragma unroll
        for (int s = 0; s < 7; ++s) acc += f[s] * sphm[blk * 8 + s];
        angle[blk * E + e] = acc;
    }
}

__global__ void k_gather_h(const int* __restrict__ Z, const float* __restrict__ emb,
                           float* __restrict__ h, int A) {
    int idx = blockIdx.x * blockDim.x + threadIdx.x;
    if (idx >= A * HID) return;
    int a = idx >> 7, c = idx & 127;
    h[idx] = emb[(Z[a] - 1) * HID + c];
}

// m = rbf @ embed_w[:, 256:384] + embed_b[256:]
__global__ void k_embed_m(const float* __restrict__ rbf, const float* __restrict__ ew,
                          const float* __restrict__ eb, float* __restrict__ m, int E) {
    int idx = blockIdx.x * blockDim.x + threadIdx.x;
    if (idx >= E * HID) return;
    int e = idx >> 7;
    int c = idx & 127;
    float acc = eb[256 + c];
#pragma unroll
    for (int r = 0; r < NR; ++r) acc += rbf[e * NR + r] * ew[r * 384 + 256 + c];
    m[idx] = acc;
}

// ---------------------------------------------------------------------------
// Edge MLP: out = silu(rbf @ w1 + b1) @ w2 + b2     (E x 128)
// ---------------------------------------------------------------------------
__global__ void __launch_bounds__(256)
k_edge_mlp(const float* __restrict__ rbf,
           const float* __restrict__ w1, const float* __restrict__ b1,
           const float* __restrict__ w2, const float* __restrict__ b2,
           float* __restrict__ out, int E) {
    extern __shared__ __align__(32) char smem[];
    float*    s_rbf = (float*)smem;                          // 64*16
    float*    s_w1  = s_rbf + 64 * NR;                       // 16*128
    _Float16* s_hid = (_Float16*)(s_w1 + NR * HID);          // 64*128
    _Float16* s_w2  = s_hid + 64 * HID;                      // 128*128 swizzled

    int tid   = threadIdx.x;
    int eBase = blockIdx.x * 64;
    bool full = (eBase + 64 <= E);

    if (full) {
        for (int idx = tid; idx < 64 * NR; idx += 256)
            s_rbf[idx] = rbf[eBase * NR + idx];
    } else {
        for (int idx = tid; idx < 64 * NR; idx += 256) {
            int row = eBase + (idx >> 4);
            s_rbf[idx] = (row < E) ? rbf[row * NR + (idx & 15)] : 0.f;
        }
    }
    for (int idx = tid; idx < NR * HID; idx += 256) s_w1[idx] = w1[idx];
    stage_w_sw(w2, s_w2, tid);
    __syncthreads();

    for (int idx = tid; idx < 64 * HID; idx += 256) {
        int e = idx >> 7, c = idx & 127;
        float acc = b1[c];
#pragma unroll
        for (int r = 0; r < NR; ++r) acc += s_rbf[e * NR + r] * s_w1[r * HID + c];
        s_hid[idx] = (_Float16)silu_f(acc);
    }
    __syncthreads();

    int wave = tid >> 5, lane = tid & 31, l16 = lane & 15, half = lane >> 4;
    int nB = wave * 16;
#pragma unroll
    for (int mi = 0; mi < 4; ++mi) {
        v8f acc = {};
#pragma unroll
        for (int kt = 0; kt < 4; ++kt) {
            v16h a = load_a(s_hid, HID, mi * 16, kt * 32, l16, half);
            v16h b = load_b_sw(s_w2, kt, wave, lane);
            acc = wmma_f16(a, b, acc);
        }
        float bias = b2[nB + l16];
        int base = (eBase + mi * 16 + 8 * half) * HID + nB + l16;
        if (full) {
#pragma unroll
            for (int r = 0; r < 8; ++r) out[base + r * HID] = acc[r] + bias;
        } else {
#pragma unroll
            for (int r = 0; r < 8; ++r) {
                int row = eBase + mi * 16 + r + 8 * half;
                if (row < E) out[row * HID + nB + l16] = acc[r] + bias;
            }
        }
    }
}

// ---------------------------------------------------------------------------
// Fused interaction: mkf = m[kj]*rad_f[kj]*sigmoid(angle[ji]) (LDS only),
// agg[dst] += mkf (atomic), m_next = m + sum_j silu(mkf @ Wj + bj)
// ---------------------------------------------------------------------------
__global__ void __launch_bounds__(256)
k_triplet(const float* __restrict__ m_cur, float* __restrict__ m_nxt,
          const float* __restrict__ radf, const float* __restrict__ angle,
          const int* __restrict__ trip, const int* __restrict__ dst,
          const float* __restrict__ w, const float* __restrict__ b,
          float* __restrict__ agg, int E) {
    extern __shared__ __align__(32) char smem[];
    float*    s_aw = (float*)smem;                 // 64
    _Float16* s_a  = (_Float16*)(s_aw + 64);       // 64*128
    _Float16* s_w  = s_a + 64 * HID;               // 128*128 swizzled

    int tid = threadIdx.x;
    int tBase = blockIdx.x * 64;
    bool full = (tBase + 64 <= E);

    if (tid < 64) {
        int tt = tBase + tid;
        float aw = 0.f;
        if (full || tt < E) aw = sigm_f(angle[trip[tt * 3 + 0]]);
        s_aw[tid] = aw;
    }
    __syncthreads();

    if (full) {
        for (int idx = tid; idx < 64 * HID; idx += 256) {
            int t = idx >> 7, c = idx & 127;
            int tt = tBase + t;
            int kj = trip[tt * 3 + 1];
            float v = m_cur[kj * HID + c] * radf[kj * HID + c] * s_aw[t];
            s_a[idx] = (_Float16)v;
            atomicAdd(&agg[dst[tt] * HID + c], v);
        }
    } else {
        for (int idx = tid; idx < 64 * HID; idx += 256) {
            int t = idx >> 7, c = idx & 127;
            int tt = tBase + t;
            float v = 0.f;
            if (tt < E) {
                int kj = trip[tt * 3 + 1];
                v = m_cur[kj * HID + c] * radf[kj * HID + c] * s_aw[t];
                atomicAdd(&agg[dst[tt] * HID + c], v);
            }
            s_a[idx] = (_Float16)v;
        }
    }

    int wave = tid >> 5, lane = tid & 31, l16 = lane & 15, half = lane >> 4;
    int nB = wave * 16;
    v8f sum[4] = {};
    for (int j = 0; j < 3; ++j) {
        __syncthreads();
        stage_w_sw(w + j * HID * HID, s_w, tid);
        __syncthreads();
#pragma unroll
        for (int mi = 0; mi < 4; ++mi) {
            v8f acc = {};
#pragma unroll
            for (int kt = 0; kt < 4; ++kt) {
                v16h a  = load_a(s_a, HID, mi * 16, kt * 32, l16, half);
                v16h bf = load_b_sw(s_w, kt, wave, lane);
                acc = wmma_f16(a, bf, acc);
            }
            float bias = b[j * HID + nB + l16];
#pragma unroll
            for (int r = 0; r < 8; ++r) sum[mi][r] += silu_f(acc[r] + bias);
        }
    }
#pragma unroll
    for (int mi = 0; mi < 4; ++mi) {
        int base = (tBase + mi * 16 + 8 * half) * HID + nB + l16;
        if (full) {
#pragma unroll
            for (int r = 0; r < 8; ++r)
                m_nxt[base + r * HID] = m_cur[base + r * HID] + sum[mi][r];
        } else {
#pragma unroll
            for (int r = 0; r < 8; ++r) {
                int row = tBase + mi * 16 + r + 8 * half;
                if (row < E) {
                    int o = row * HID + nB + l16;
                    m_nxt[o] = m_cur[o] + sum[mi][r];
                }
            }
        }
    }
}

// ---------------------------------------------------------------------------
// Atom update: h += silu([h,agg] @ w1 + b1) @ w2 + b2   (A x 128, K=256 then 128)
// ---------------------------------------------------------------------------
__global__ void __launch_bounds__(256)
k_atom_update(float* __restrict__ h, const float* __restrict__ agg,
              const float* __restrict__ w1, const float* __restrict__ b1,
              const float* __restrict__ w2, const float* __restrict__ b2,
              int A) {
    extern __shared__ __align__(32) char smem[];
    _Float16* s_c = (_Float16*)smem;       // 64*256 comb; later 64*128 hidden
    _Float16* s_w = s_c + 64 * 256;        // 128*128 swizzled

    int tid = threadIdx.x;
    int aBase = blockIdx.x * 64;
    bool full = (aBase + 64 <= A);

    if (full) {
        for (int idx = tid; idx < 64 * 256; idx += 256) {
            int r = idx >> 8, k = idx & 255;
            int row = aBase + r;
            float v = (k < HID) ? h[row * HID + k] : agg[row * HID + (k - HID)];
            s_c[idx] = (_Float16)v;
        }
    } else {
        for (int idx = tid; idx < 64 * 256; idx += 256) {
            int r = idx >> 8, k = idx & 255;
            int row = aBase + r;
            float v = 0.f;
            if (row < A) v = (k < HID) ? h[row * HID + k] : agg[row * HID + (k - HID)];
            s_c[idx] = (_Float16)v;
        }
    }

    int wave = tid >> 5, lane = tid & 31, l16 = lane & 15, half = lane >> 4;
    int nB = wave * 16;
    v8f accs[4] = {};
    for (int kh = 0; kh < 2; ++kh) {
        __syncthreads();
        stage_w_sw(w1 + kh * HID * HID, s_w, tid);
        __syncthreads();
#pragma unroll
        for (int mi = 0; mi < 4; ++mi)
#pragma unroll
            for (int kt = 0; kt < 4; ++kt) {
                v16h a = load_a(s_c, 256, mi * 16, kh * HID + kt * 32, l16, half);
                v16h b = load_b_sw(s_w, kt, wave, lane);
                accs[mi] = wmma_f16(a, b, accs[mi]);
            }
    }
    __syncthreads();                       // comb fully consumed
    _Float16* s_h = s_c;                   // reuse as hidden 64x128
#pragma unroll
    for (int mi = 0; mi < 4; ++mi) {
        float bias = b1[nB + l16];
#pragma unroll
        for (int r = 0; r < 8; ++r) {
            float v = accs[mi][r] + bias;
            s_h[(mi * 16 + r + 8 * half) * HID + nB + l16] = (_Float16)silu_f(v);
        }
    }
    __syncthreads();
    stage_w_sw(w2, s_w, tid);
    __syncthreads();
#pragma unroll
    for (int mi = 0; mi < 4; ++mi) {
        v8f acc = {};
#pragma unroll
        for (int kt = 0; kt < 4; ++kt) {
            v16h a = load_a(s_h, HID, mi * 16, kt * 32, l16, half);
            v16h b = load_b_sw(s_w, kt, wave, lane);
            acc = wmma_f16(a, b, acc);
        }
        float bias = b2[nB + l16];
        int base = (aBase + mi * 16 + 8 * half) * HID + nB + l16;
        if (full) {
#pragma unroll
            for (int r = 0; r < 8; ++r) h[base + r * HID] += acc[r] + bias;
        } else {
#pragma unroll
            for (int r = 0; r < 8; ++r) {
                int row = aBase + mi * 16 + r + 8 * half;
                if (row < A) h[row * HID + nB + l16] += acc[r] + bias;
            }
        }
    }
}

// ---------------------------------------------------------------------------
// Output-block scatter: x[dst[e]] += h[src[e]] * W[e]
// ---------------------------------------------------------------------------
__global__ void k_ob_scatter(const float* __restrict__ W, const float* __restrict__ h,
                             const int* __restrict__ src, const int* __restrict__ dst,
                             float* __restrict__ x, int E) {
    int idx = blockIdx.x * blockDim.x + threadIdx.x;
    if (idx >= E * HID) return;
    int e = idx >> 7;
    int c = idx & 127;
    float v = h[src[e] * HID + c] * W[idx];
    atomicAdd(&x[dst[e] * HID + c], v);
}

// ---------------------------------------------------------------------------
// Output-block dense chain: x = silu(x@Wj+bj) x3, then out += x @ wo + bo
// ---------------------------------------------------------------------------
__global__ void __launch_bounds__(256)
k_ob_dense(const float* __restrict__ x,
           const float* __restrict__ wd, const float* __restrict__ bd,
           const float* __restrict__ wo, const float* __restrict__ bo,
           float* __restrict__ out, int A) {
    extern __shared__ __align__(32) char smem[];
    _Float16* s_x0 = (_Float16*)smem;      // 64*128
    _Float16* s_x1 = s_x0 + 64 * HID;      // 64*128
    _Float16* s_w  = s_x1 + 64 * HID;      // 128*128 swizzled

    int tid = threadIdx.x;
    int aBase = blockIdx.x * 64;
    bool full = (aBase + 64 <= A);

    if (full) {
        for (int idx = tid; idx < 64 * HID; idx += 256)
            s_x0[idx] = (_Float16)x[aBase * HID + idx];
    } else {
        for (int idx = tid; idx < 64 * HID; idx += 256) {
            int row = aBase + (idx >> 7);
            s_x0[idx] = (_Float16)((row < A) ? x[row * HID + (idx & 127)] : 0.f);
        }
    }

    _Float16* cur = s_x0;
    _Float16* nxt = s_x1;
    int wave = tid >> 5, lane = tid & 31, l16 = lane & 15, half = lane >> 4;
    int nB = wave * 16;
    for (int j = 0; j < 3; ++j) {
        stage_w_sw(wd + j * HID * HID, s_w, tid);
        __syncthreads();
#pragma unroll
        for (int mi = 0; mi < 4; ++mi) {
            v8f acc = {};
#pragma unroll
            for (int kt = 0; kt < 4; ++kt) {
                v16h a = load_a(cur, HID, mi * 16, kt * 32, l16, half);
                v16h b = load_b_sw(s_w, kt, wave, lane);
                acc = wmma_f16(a, b, acc);
            }
            float bias = bd[j * HID + nB + l16];
#pragma unroll
            for (int r = 0; r < 8; ++r) {
                float v = acc[r] + bias;
                nxt[(mi * 16 + r + 8 * half) * HID + nB + l16] = (_Float16)silu_f(v);
            }
        }
        __syncthreads();
        _Float16* t = cur; cur = nxt; nxt = t;
    }
    if (tid < 64) {
        int row = aBase + tid;
        if (row < A) {
            float acc = bo[0];
#pragma unroll 8
            for (int c = 0; c < HID; ++c) acc += (float)cur[tid * HID + c] * wo[c];
            out[row] += acc;
        }
    }
}

// ---------------------------------------------------------------------------
// Host launch
// ---------------------------------------------------------------------------
extern "C" void kernel_launch(void* const* d_in, const int* in_sizes, int n_in,
                              void* d_out, int out_size, void* d_ws, size_t ws_size,
                              hipStream_t stream) {
    const int*   Z        = (const int*)d_in[0];
    const int*   eidx     = (const int*)d_in[1];
    const float* dist     = (const float*)d_in[2];
    const float* dirs     = (const float*)d_in[3];
    const int*   trip     = (const int*)d_in[4];
    const float* atom_emb = (const float*)d_in[5];
    const float* embed_w  = (const float*)d_in[6];
    const float* embed_b  = (const float*)d_in[7];
    const float* ib_rad_w1 = (const float*)d_in[8];
    const float* ib_rad_b1 = (const float*)d_in[9];
    const float* ib_rad_w2 = (const float*)d_in[10];
    const float* ib_rad_b2 = (const float*)d_in[11];
    const float* ib_sph_w  = (const float*)d_in[12];
    const float* ib_sph_b  = (const float*)d_in[13];
    const float* ib_upd_w1 = (const float*)d_in[14];
    const float* ib_upd_b1 = (const float*)d_in[15];
    const float* ib_upd_w2 = (const float*)d_in[16];
    const float* ib_upd_b2 = (const float*)d_in[17];
    const float* ib_out_w  = (const float*)d_in[18];
    const float* ib_out_b  = (const float*)d_in[19];
    const float* ob_rad_w1 = (const float*)d_in[20];
    const float* ob_rad_b1 = (const float*)d_in[21];
    const float* ob_rad_w2 = (const float*)d_in[22];
    const float* ob_rad_b2 = (const float*)d_in[23];
    const float* ob_dense_w = (const float*)d_in[24];
    const float* ob_dense_b = (const float*)d_in[25];
    const float* ob_out_w   = (const float*)d_in[26];
    const float* ob_out_b   = (const float*)d_in[27];

    const int E = in_sizes[2];        // edges (== triplets here)
    const int A = in_sizes[0];        // atoms
    const int* src = eidx;
    const int* dst = eidx + E;

    // workspace layout (fp32)
    float* ws   = (float*)d_ws;
    float* rbf   = ws;                         size_t o = (size_t)E * NR;
    float* angle = ws + o;                     o += (size_t)4 * E;
    float* h     = ws + o;                     o += (size_t)A * HID;
    float* mA    = ws + o;                     o += (size_t)E * HID;
    float* mB    = ws + o;                     o += (size_t)E * HID;
    float* radf  = ws + o;                     o += (size_t)E * HID;
    float* agg   = ws + o;                     o += (size_t)A * HID;
    float* xbuf  = ws + o;                     o += (size_t)A * HID;
    float* sphm  = ws + o;                     o += 32;
    (void)o; (void)ws_size; (void)n_in;

    const int TPB = 256;
    const int gE    = (E + TPB - 1) / TPB;
    const int gEH   = (int)(((size_t)E * HID + TPB - 1) / TPB);
    const int gAH   = (A * HID + TPB - 1) / TPB;
    const int gTile = (E + 63) / 64;
    const int gAT   = (A + 63) / 64;

    const size_t SM_EDGE = 64 * NR * 4 + NR * HID * 4 + 64 * HID * 2 + HID * HID * 2; // 60 KB
    const size_t SM_TRIP = 64 * 4 + 64 * HID * 2 + HID * HID * 2;                      // 49 KB
    const size_t SM_ATOM = 64 * 256 * 2 + HID * HID * 2;                               // 64 KB
    const size_t SM_OBD  = 2 * 64 * HID * 2 + HID * HID * 2;                           // 64 KB

    hipMemsetAsync(d_out, 0, (size_t)out_size * sizeof(float), stream);

    // --- setup ---
    k_rbf<<<gE, TPB, 0, stream>>>(dist, rbf, E);
    k_sphmean<<<1, 32, 0, stream>>>(ib_sph_w, ib_sph_b, sphm);
    k_angle<<<gE, TPB, 0, stream>>>(dirs, sphm, angle, E);
    k_gather_h<<<gAH, TPB, 0, stream>>>(Z, atom_emb, h, A);
    k_embed_m<<<gEH, TPB, 0, stream>>>(rbf, embed_w, embed_b, mA, E);

    float* m_cur = mA;
    float* m_nxt = mB;

    // --- output block 0 (on initial h) ---
    k_edge_mlp<<<gTile, TPB, SM_EDGE, stream>>>(rbf, ob_rad_w1, ob_rad_b1,
                                                ob_rad_w2, ob_rad_b2, radf, E);
    hipMemsetAsync(xbuf, 0, (size_t)A * HID * sizeof(float), stream);
    k_ob_scatter<<<gEH, TPB, 0, stream>>>(radf, h, src, dst, xbuf, E);
    k_ob_dense<<<gAT, TPB, SM_OBD, stream>>>(xbuf, ob_dense_w, ob_dense_b,
                                             ob_out_w, ob_out_b, (float*)d_out, A);

    // --- interaction blocks ---
    for (int i = 0; i < 4; ++i) {
        k_edge_mlp<<<gTile, TPB, SM_EDGE, stream>>>(
            rbf, ib_rad_w1 + (size_t)i * NR * HID, ib_rad_b1 + (size_t)i * HID,
            ib_rad_w2 + (size_t)i * HID * HID, ib_rad_b2 + (size_t)i * HID, radf, E);

        hipMemsetAsync(agg, 0, (size_t)A * HID * sizeof(float), stream);
        k_triplet<<<gTile, TPB, SM_TRIP, stream>>>(
            m_cur, m_nxt, radf, angle + (size_t)i * E, trip, dst,
            ib_out_w + (size_t)i * 3 * HID * HID, ib_out_b + (size_t)i * 3 * HID, agg, E);
        { float* t = m_cur; m_cur = m_nxt; m_nxt = t; }

        k_atom_update<<<gAT, TPB, SM_ATOM, stream>>>(
            h, agg, ib_upd_w1 + (size_t)i * 2 * HID * HID, ib_upd_b1 + (size_t)i * HID,
            ib_upd_w2 + (size_t)i * HID * HID, ib_upd_b2 + (size_t)i * HID, A);

        // output block i+1
        int ob = i + 1;
        k_edge_mlp<<<gTile, TPB, SM_EDGE, stream>>>(
            rbf, ob_rad_w1 + (size_t)ob * NR * HID, ob_rad_b1 + (size_t)ob * HID,
            ob_rad_w2 + (size_t)ob * HID * HID, ob_rad_b2 + (size_t)ob * HID, radf, E);
        hipMemsetAsync(xbuf, 0, (size_t)A * HID * sizeof(float), stream);
        k_ob_scatter<<<gEH, TPB, 0, stream>>>(radf, h, src, dst, xbuf, E);
        k_ob_dense<<<gAT, TPB, SM_OBD, stream>>>(
            xbuf, ob_dense_w + (size_t)ob * 3 * HID * HID, ob_dense_b + (size_t)ob * 3 * HID,
            ob_out_w + (size_t)ob * HID, ob_out_b + ob, (float*)d_out, A);
    }
}
